// CustomLSTM_74663711473832
// MI455X (gfx1250) — matrix operand order repaired and checked by reading, not compiled, not run on hardware
//
#include <hip/hip_runtime.h>
#include <hip/hip_bf16.h>

typedef _Float16 half_t;
typedef __attribute__((ext_vector_type(16))) _Float16 v16h;
typedef __attribute__((ext_vector_type(8)))  _Float16 h8;
typedef __attribute__((ext_vector_type(8)))  float    v8f;

union V16U { v16h v; h8 h[2]; };

#define BATCH 512
#define TSEQ  256
#define HID   64
#define G4    256          // 4*HID gate columns
#define MROW  (BATCH*TSEQ) // 131072 rows for the time-parallel GEMM
#define K1PAD 288          // 258 padded up to multiple of 32

__device__ __forceinline__ v8f wmma16(v16h a, v16h b, v8f c) {
    return __builtin_amdgcn_wmma_f32_16x16x32_f16(
        /*neg_a=*/false, a, /*neg_b=*/false, b,
        /*c_mod=*/(short)0, c, /*reuse_a=*/false, /*reuse_b=*/false);
}

// A tile (16x32, 16-bit): lane m=L%16; K runs [kb+half*8, +8) and [kb+16+half*8, +8)
__device__ __forceinline__ v16h load_a16(const half_t* __restrict__ A, int lda,
                                         int m, int half, int kb) {
    V16U u;
    const half_t* p = A + (size_t)m * lda + kb + half * 8;
    u.h[0] = *(const h8*)(p);
    u.h[1] = *(const h8*)(p + 16);
    return u.v;
}

// B tile (32x16, 16-bit) from row-major W[n][k]: lane n=L%16; K runs [kb+half*8, +8) and [kb+16+half*8, +8)
__device__ __forceinline__ v16h load_b16(const half_t* __restrict__ W, int ldw,
                                         int n, int half, int kb) {
    V16U u;
    const half_t* p = W + (size_t)n * ldw + kb + half * 8;
    u.h[0] = *(const h8*)(p);
    u.h[1] = *(const h8*)(p + 16);
    return u.v;
}

// ---------------- fp32 -> f16 convert with optional column zero-pad ----------------
__global__ void pad_convert_kernel(const float* __restrict__ src, half_t* __restrict__ dst,
                                   unsigned rows, unsigned sk, unsigned dk) {
    unsigned i = blockIdx.x * 256u + threadIdx.x;
    unsigned n = rows * dk;
    if (i >= n) return;
    unsigned r = i / dk, k = i % dk;
    dst[i] = (k < sk) ? (half_t)src[(size_t)r * sk + k] : (half_t)0.f;
}

// ---------------- time-parallel gate GEMM: xg = A @ W^T + bih + bhh ----------------
// A: (MROW x K) f16 row-major, W: (256 x K) f16 row-major, xg: (MROW x 256) fp32
// grid = MROW/16 blocks, 256 threads (8 waves); wave w owns N-tiles 2w, 2w+1
__global__ void xgemm_kernel(const half_t* __restrict__ A, const half_t* __restrict__ W,
                             const float* __restrict__ bih, const float* __restrict__ bhh,
                             float* __restrict__ xg, int K) {
    int tid  = threadIdx.x;
    int lane = tid & 31, w = tid >> 5;
    int m16 = lane & 15, half = lane >> 4;
    int mrow = blockIdx.x * 16;
    int n0 = (2 * w) * 16 + m16;
    int n1 = n0 + 16;

    v8f acc0 = {}, acc1 = {};
    const half_t* Ab = A + (size_t)mrow * K;
    for (int kb = 0; kb < K; kb += 32) {
        v16h a  = load_a16(Ab, K, m16, half, kb);
        v16h b0 = load_b16(W, K, n0, half, kb);
        v16h b1 = load_b16(W, K, n1, half, kb);
        acc0 = wmma16(a, b0, acc0);
        acc1 = wmma16(a, b1, acc1);
    }
    float bias0 = bih[n0] + bhh[n0];
    float bias1 = bih[n1] + bhh[n1];
#pragma unroll
    for (int v = 0; v < 8; v++) {
        int m = v + 8 * half;
        xg[(size_t)(mrow + m) * G4 + n0] = acc0[v] + bias0;
        xg[(size_t)(mrow + m) * G4 + n1] = acc1[v] + bias1;
    }
}

// ---------------- recurrence: one block = 16 batch rows, all T steps ----------------
// xg: (B*T x 256) fp32 with row b*T+t; Whh: (256 x 64) f16; hout: (B*T x 64) f16
// xg loads for step t+1 are issued during step t (register double-buffer) so the
// serial chain only sees LDS + barrier latency, not HBM latency.
__global__ void lstm_rec_kernel(const float* __restrict__ xg, const half_t* __restrict__ Whh,
                                half_t* __restrict__ hout, int T) {
    __shared__ float gsh[16 * G4];
    __shared__ float csh[16 * HID];
    __shared__ alignas(16) half_t hsh[16 * HID];

    int tid  = threadIdx.x;
    int lane = tid & 31, w = tid >> 5;
    int m16 = lane & 15, half = lane >> 4;
    int rowbase = blockIdx.x * 16;

    // preload Whh B tiles: [n-tile j][k-tile kt]
    v16h B[2][2];
#pragma unroll
    for (int j = 0; j < 2; j++) {
        int n = (2 * w + j) * 16 + m16;
#pragma unroll
        for (int kt = 0; kt < 2; kt++) B[j][kt] = load_b16(Whh, HID, n, half, kt * 32);
    }
    for (int i = tid; i < 16 * HID; i += 256) { csh[i] = 0.f; hsh[i] = (half_t)0.f; }

    auto load_x = [&](int t, float v[4][4]) {
#pragma unroll
        for (int j = 0; j < 4; j++) {
            int p = j * 256 + tid;
            int hid = p & (HID - 1), m = p >> 6;
            size_t xoff = ((size_t)(rowbase + m) * T + t) * G4 + hid;
            v[j][0] = xg[xoff];
            v[j][1] = xg[xoff + HID];
            v[j][2] = xg[xoff + 2 * HID];
            v[j][3] = xg[xoff + 3 * HID];
        }
    };

    float xv[4][4];
    load_x(0, xv);           // prefetch step 0
    __syncthreads();

    for (int t = 0; t < T; t++) {
        // ---- matrix phase: gsh = h @ Whh^T ----
        v16h a0 = load_a16(hsh, HID, m16, half, 0);
        v16h a1 = load_a16(hsh, HID, m16, half, 32);
#pragma unroll
        for (int j = 0; j < 2; j++) {
            v8f acc = {};
            acc = wmma16(a0, B[j][0], acc);
            acc = wmma16(a1, B[j][1], acc);
            int col = (2 * w + j) * 16 + m16;
#pragma unroll
            for (int v = 0; v < 8; v++) gsh[(v + 8 * half) * G4 + col] = acc[v];
        }
        // issue next step's xg loads now; consumed after two barriers
        float xn[4][4];
        load_x((t + 1 < T) ? t + 1 : t, xn);
        __syncthreads();
        // ---- cell phase: 256 threads x 4 (m,hid) pairs ----
#pragma unroll
        for (int j = 0; j < 4; j++) {
            int p = j * 256 + tid;
            int hid = p & (HID - 1), m = p >> 6;
            float gi = gsh[m * G4 + hid]           + xv[j][0];
            float gf = gsh[m * G4 + HID + hid]     + xv[j][1];
            float gg = gsh[m * G4 + 2 * HID + hid] + xv[j][2];
            float go = gsh[m * G4 + 3 * HID + hid] + xv[j][3];
            gi = 1.f / (1.f + __expf(-gi));
            gf = 1.f / (1.f + __expf(-gf));
            gg = tanhf(gg);
            go = 1.f / (1.f + __expf(-go));
            float c = gf * csh[m * HID + hid] + gi * gg;
            float h = go * tanhf(c);
            csh[m * HID + hid] = c;
            hsh[m * HID + hid] = (half_t)h;
            hout[((size_t)(rowbase + m) * T + t) * HID + hid] = (half_t)h;
        }
#pragma unroll
        for (int j = 0; j < 4; j++)
#pragma unroll
            for (int q = 0; q < 4; q++) xv[j][q] = xn[j][q];
        __syncthreads();
    }
}

// ---------------- head helpers ----------------
__global__ void extract_last_kernel(const half_t* __restrict__ hbuf, float* __restrict__ z, int T) {
    int i = blockIdx.x * 256 + threadIdx.x;
    if (i >= BATCH * HID) return;
    int b = i >> 6, k = i & 63;
    z[i] = (float)hbuf[((size_t)b * T + (T - 1)) * HID + k];
}

__global__ void fc_kernel(const float* __restrict__ in, const float* __restrict__ W,
                          const float* __restrict__ b, float* __restrict__ out,
                          int M, int N, int K, int relu) {
    int i = blockIdx.x * 256 + threadIdx.x;
    if (i >= M * N) return;
    int r = i / N, n = i % N;
    float s = b[n];
    for (int k = 0; k < K; k++) s += in[r * K + k] * W[n * K + k];
    if (relu) s = fmaxf(s, 0.f);
    out[i] = s;
}

extern "C" void kernel_launch(void* const* d_in, const int* in_sizes, int n_in,
                              void* d_out, int out_size, void* d_ws, size_t ws_size,
                              hipStream_t stream) {
    const float* x    = (const float*)d_in[0];
    const float* Wih[3] = {(const float*)d_in[1], (const float*)d_in[5], (const float*)d_in[9]};
    const float* Whh[3] = {(const float*)d_in[2], (const float*)d_in[6], (const float*)d_in[10]};
    const float* bih[3] = {(const float*)d_in[3], (const float*)d_in[7], (const float*)d_in[11]};
    const float* bhh[3] = {(const float*)d_in[4], (const float*)d_in[8], (const float*)d_in[12]};
    const float* Wfc[5] = {(const float*)d_in[13], (const float*)d_in[15], (const float*)d_in[17],
                           (const float*)d_in[19], (const float*)d_in[21]};
    const float* bfc[5] = {(const float*)d_in[14], (const float*)d_in[16], (const float*)d_in[18],
                           (const float*)d_in[20], (const float*)d_in[22]};
    const float* Wout = (const float*)d_in[23];
    const float* bout = (const float*)d_in[24];
    float* out = (float*)d_out;

    char* ws = (char*)d_ws;
    size_t off = 0;
    auto take = [&](size_t bytes) { char* p = ws + off; off += (bytes + 255) & ~(size_t)255; return p; };
    float*  xg     = (float*)take((size_t)MROW * G4 * 4);     // 128 MB
    half_t* xA     = (half_t*)take((size_t)MROW * K1PAD * 2); // 75.5 MB
    half_t* hbuf   = (half_t*)take((size_t)MROW * HID * 2);   // 16 MB
    half_t* wih1p  = (half_t*)take((size_t)G4 * K1PAD * 2);
    half_t* wihh[2]; wihh[0] = (half_t*)take((size_t)G4 * HID * 2);
                     wihh[1] = (half_t*)take((size_t)G4 * HID * 2);
    half_t* whhh[3]; for (int l = 0; l < 3; l++) whhh[l] = (half_t*)take((size_t)G4 * HID * 2);
    float* zA = (float*)take((size_t)BATCH * 128 * 4);
    float* zB = (float*)take((size_t)BATCH * 128 * 4);

    auto cgrid = [](size_t n) { return (unsigned)((n + 255) / 256); };

    // convert/pad weights + input to f16
    pad_convert_kernel<<<cgrid((size_t)MROW * K1PAD), 256, 0, stream>>>(x, xA, MROW, 258, K1PAD);
    pad_convert_kernel<<<cgrid((size_t)G4 * K1PAD), 256, 0, stream>>>(Wih[0], wih1p, G4, 258, K1PAD);
    pad_convert_kernel<<<cgrid((size_t)G4 * HID), 256, 0, stream>>>(Wih[1], wihh[0], G4, HID, HID);
    pad_convert_kernel<<<cgrid((size_t)G4 * HID), 256, 0, stream>>>(Wih[2], wihh[1], G4, HID, HID);
    for (int l = 0; l < 3; l++)
        pad_convert_kernel<<<cgrid((size_t)G4 * HID), 256, 0, stream>>>(Whh[l], whhh[l], G4, HID, HID);

    // layer 1
    xgemm_kernel<<<MROW / 16, 256, 0, stream>>>(xA, wih1p, bih[0], bhh[0], xg, K1PAD);
    lstm_rec_kernel<<<BATCH / 16, 256, 0, stream>>>(xg, whhh[0], hbuf, TSEQ);
    // layer 2
    xgemm_kernel<<<MROW / 16, 256, 0, stream>>>(hbuf, wihh[0], bih[1], bhh[1], xg, HID);
    lstm_rec_kernel<<<BATCH / 16, 256, 0, stream>>>(xg, whhh[1], hbuf, TSEQ);
    // layer 3
    xgemm_kernel<<<MROW / 16, 256, 0, stream>>>(hbuf, wihh[1], bih[2], bhh[2], xg, HID);
    lstm_rec_kernel<<<BATCH / 16, 256, 0, stream>>>(xg, whhh[2], hbuf, TSEQ);

    // head
    extract_last_kernel<<<cgrid(BATCH * HID), 256, 0, stream>>>(hbuf, zA, TSEQ);
    fc_kernel<<<cgrid(512 * 64),  256, 0, stream>>>(zA, Wfc[0], bfc[0], zB, 512, 64, 64, 1);
    fc_kernel<<<cgrid(512 * 128), 256, 0, stream>>>(zB, Wfc[1], bfc[1], zA, 512, 128, 64, 1);
    fc_kernel<<<cgrid(512 * 64),  256, 0, stream>>>(zA, Wfc[2], bfc[2], zB, 512, 64, 128, 1);
    fc_kernel<<<cgrid(512 * 32),  256, 0, stream>>>(zB, Wfc[3], bfc[3], zA, 512, 32, 64, 1);
    fc_kernel<<<cgrid(512 * 32),  256, 0, stream>>>(zA, Wfc[4], bfc[4], zB, 512, 32, 32, 1);
    fc_kernel<<<cgrid(512 * 90),  256, 0, stream>>>(zB, Wout, bout, out, 512, 90, 32, 0);
}
